// VQEmbedding_43267500540460
// MI455X (gfx1250) — compile-verified, matrix operand order checked by reading, not run on hardware
//
#include <hip/hip_runtime.h>
#include <hip/hip_bf16.h>

typedef __attribute__((ext_vector_type(8)))  _Float16 v8h;
typedef __attribute__((ext_vector_type(16))) _Float16 v16h;
typedef __attribute__((ext_vector_type(8)))  float    v8f;

#define DDIM   256
#define NROWS  32768      // 8 * 4096
#define MCODES 1024
#define COMMIT 0.25f
#define EPSV   1e-10f
#define WAVES_PER_BLOCK 8
#define KSTEPS (DDIM / 32) // 8 WMMA K-steps of 32 (f16 path)

// Build a v16h A/B fragment from the two contiguous 8-element runs the
// 16-bit WMMA layout expects: elements 0-7 <- p0[0..7], 8-15 <- p1[0..7].
static __device__ __forceinline__ v16h make_frag(const _Float16* p0, const _Float16* p1) {
  v8h a = *(const v8h*)p0;
  v8h b = *(const v8h*)p1;
  return __builtin_shufflevector(a, b, 0, 1, 2, 3, 4, 5, 6, 7,
                                 8, 9, 10, 11, 12, 13, 14, 15);
}

// Split 8 contiguous f32 into f16 hi + f16 lo (error-corrected split).
static __device__ __forceinline__ void cvt8(const float* __restrict__ p, v8h& hi, v8h& lo) {
#pragma unroll
  for (int j = 0; j < 8; ++j) {
    float v = p[j];
    _Float16 h = (_Float16)v;
    hi[j] = h;
    lo[j] = (_Float16)(v - (float)h);
  }
}

// ---------------- codebook squared norms (exact f32) ----------------
__global__ __launch_bounds__(256)
void enorm_kernel(const float* __restrict__ emb, float* __restrict__ enorm) {
  int m = blockIdx.x * blockDim.x + threadIdx.x;
  if (m >= MCODES) return;
  const float4* p = (const float4*)(emb + (size_t)m * DDIM);
  float s = 0.f;
#pragma unroll 8
  for (int i = 0; i < DDIM / 4; ++i) {
    float4 v = p[i];
    s += v.x * v.x + v.y * v.y + v.z * v.z + v.w * v.w;
  }
  enorm[m] = s;
}

// ---------------- codebook f32 -> f16 hi/lo split ----------------
__global__ __launch_bounds__(256)
void emb_cvt_kernel(const float* __restrict__ emb,
                    _Float16* __restrict__ ehi, _Float16* __restrict__ elo) {
  int i = blockIdx.x * blockDim.x + threadIdx.x;   // over MCODES*DDIM
  float v = emb[i];
  _Float16 h = (_Float16)v;
  ehi[i] = h;
  elo[i] = (_Float16)(v - (float)h);
}

// ---------------- zero accumulators (deterministic per call) ----------------
__global__ __launch_bounds__(256)
void init_kernel(float* __restrict__ counts, float* __restrict__ sumsq) {
  int t = blockIdx.x * blockDim.x + threadIdx.x;
  if (t < MCODES) counts[t] = 0.f;
  if (t == 0) *sumsq = 0.f;
}

// ---------------- fused distance GEMM + argmin ----------------
// Split-FP16 error-corrected GEMM on v_wmma_f32_16x16x32_f16:
//   dot(x,e) ~= xh*eh + xh*el + xl*eh   (error ~2^-22, argmin-safe)
// A (x tile) hi/lo fragments resident in VGPRs; 2 code tiles in flight.
__global__ __launch_bounds__(32 * WAVES_PER_BLOCK)
void vq_argmin_kernel(const float* __restrict__ x,
                      const _Float16* __restrict__ ehi,
                      const _Float16* __restrict__ elo,
                      const float* __restrict__ enorm,
                      int* __restrict__ idx_i, float* __restrict__ idx_f) {
  const int lane = threadIdx.x & 31;
  const int wave = threadIdx.x >> 5;
  const int rowTile = blockIdx.x * WAVES_PER_BLOCK + wave;   // 0..2047
  const int row0 = rowTile * 16;
  const int half = lane >> 4;
  const int l16  = lane & 15;

  // 16-bit A 16x32 layout: lane holds row=row0+l16,
  // runs K=[32k+8*half, +8) in VGPR0-3 and K=[32k+16+8*half, +8) in VGPR4-7.
  const float* aRow = x + (size_t)(row0 + l16) * DDIM;

  v16h ahi[KSTEPS], alo[KSTEPS];
#pragma unroll
  for (int k = 0; k < KSTEPS; ++k) {
    v8h h0, l0, h1, l1;
    cvt8(aRow + 32 * k + 8 * half, h0, l0);
    cvt8(aRow + 32 * k + 16 + 8 * half, h1, l1);
    ahi[k] = __builtin_shufflevector(h0, h1, 0, 1, 2, 3, 4, 5, 6, 7,
                                     8, 9, 10, 11, 12, 13, 14, 15);
    alo[k] = __builtin_shufflevector(l0, l1, 0, 1, 2, 3, 4, 5, 6, 7,
                                     8, 9, 10, 11, 12, 13, 14, 15);
  }

  float bestVal[8];
  int   bestIdx[8];
#pragma unroll
  for (int i = 0; i < 8; ++i) { bestVal[i] = 3.4028235e38f; bestIdx[i] = 0; }

  for (int ct = 0; ct < MCODES / 32; ++ct) {               // 2 code tiles per pass
    const int col0 = (ct * 2) * 16 + l16;                  // candidate codes for this lane
    const int col1 = col0 + 16;
    const _Float16* bh0 = ehi + (size_t)col0 * DDIM;
    const _Float16* bl0 = elo + (size_t)col0 * DDIM;
    const _Float16* bh1 = ehi + (size_t)col1 * DDIM;
    const _Float16* bl1 = elo + (size_t)col1 * DDIM;

    v8f c0 = {};
    v8f c1 = {};
#pragma unroll
    for (int k = 0; k < KSTEPS; ++k) {
      const int o0 = 32 * k + 8 * half;
      const int o1 = 32 * k + 16 + 8 * half;
      v16h Bh0 = make_frag(bh0 + o0, bh0 + o1);
      v16h Bl0 = make_frag(bl0 + o0, bl0 + o1);
      v16h Bh1 = make_frag(bh1 + o0, bh1 + o1);
      v16h Bl1 = make_frag(bl1 + o0, bl1 + o1);
      c0 = __builtin_amdgcn_wmma_f32_16x16x32_f16(false, ahi[k], false, Bh0,
                                                  (short)0, c0, false, false);
      c1 = __builtin_amdgcn_wmma_f32_16x16x32_f16(false, ahi[k], false, Bh1,
                                                  (short)0, c1, false, false);
      c0 = __builtin_amdgcn_wmma_f32_16x16x32_f16(false, ahi[k], false, Bl0,
                                                  (short)0, c0, false, false);
      c1 = __builtin_amdgcn_wmma_f32_16x16x32_f16(false, ahi[k], false, Bl1,
                                                  (short)0, c1, false, false);
      c0 = __builtin_amdgcn_wmma_f32_16x16x32_f16(false, alo[k], false, Bh0,
                                                  (short)0, c0, false, false);
      c1 = __builtin_amdgcn_wmma_f32_16x16x32_f16(false, alo[k], false, Bh1,
                                                  (short)0, c1, false, false);
    }

    const float en0 = enorm[col0];
    const float en1 = enorm[col1];
#pragma unroll
    for (int i = 0; i < 8; ++i) {
      // C/D layout: VGPR i, lanes 0-15 -> row i, lanes 16-31 -> row 8+i; N = lane%16
      float d0 = en0 - 2.f * c0[i];
      if (d0 < bestVal[i]) { bestVal[i] = d0; bestIdx[i] = col0; }
      float d1 = en1 - 2.f * c1[i];
      if (d1 < bestVal[i]) { bestVal[i] = d1; bestIdx[i] = col1; }
    }
  }

  // per-row argmin: reduce across the 16 lanes of each half, ties -> lower index
#pragma unroll
  for (int i = 0; i < 8; ++i) {
    float v = bestVal[i];
    int   id = bestIdx[i];
#pragma unroll
    for (int off = 8; off >= 1; off >>= 1) {
      float v2  = __shfl_xor(v, off, 16);
      int   id2 = __shfl_xor(id, off, 16);
      if (v2 < v || (v2 == v && id2 < id)) { v = v2; id = id2; }
    }
    if (l16 == 0) {
      int r = row0 + i + 8 * half;
      idx_i[r] = id;
      idx_f[r] = (float)id;
    }
  }
}

// ---------------- gather, straight-through output, loss partials, counts ----------------
__global__ __launch_bounds__(256)
void vq_gather_kernel(const float* __restrict__ x, const float* __restrict__ emb,
                      const int* __restrict__ idx,
                      float* __restrict__ qst, float* __restrict__ counts,
                      float* __restrict__ sumsq) {
  __shared__ float wsum[8];
  const int t64 = threadIdx.x & 63;       // 64 threads x float4 cover D=256
  const int rIn = threadIdx.x >> 6;       // 4 rows per block
  const int row = blockIdx.x * 4 + rIn;
  const int id = idx[row];
  const float4* xp = (const float4*)(x + (size_t)row * DDIM);
  const float4* ep = (const float4*)(emb + (size_t)id * DDIM);
  float4 xv = xp[t64];
  float4 ev = ep[t64];
  float4 q;  // forward value of x + stop_grad(q - x), computed the same way
  q.x = xv.x + (ev.x - xv.x);
  q.y = xv.y + (ev.y - xv.y);
  q.z = xv.z + (ev.z - xv.z);
  q.w = xv.w + (ev.w - xv.w);
  ((float4*)(qst + (size_t)row * DDIM))[t64] = q;

  float dx = xv.x - ev.x, dy = xv.y - ev.y, dz = xv.z - ev.z, dw = xv.w - ev.w;
  float s = dx * dx + dy * dy + dz * dz + dw * dw;
#pragma unroll
  for (int off = 16; off >= 1; off >>= 1) s += __shfl_xor(s, off, 32);
  const int lane = threadIdx.x & 31, wv = threadIdx.x >> 5;
  if (lane == 0) wsum[wv] = s;
  __syncthreads();
  if (threadIdx.x == 0) {
    float tot = 0.f;
#pragma unroll
    for (int i = 0; i < 8; ++i) tot += wsum[i];
    atomicAdd(sumsq, tot);
  }
  if (t64 == 0) atomicAdd(&counts[id], 1.0f);
}

// ---------------- scalars: loss + perplexity ----------------
__global__ __launch_bounds__(256)
void vq_finalize_kernel(const float* __restrict__ counts, const float* __restrict__ sumsq,
                        float* __restrict__ out_loss, float* __restrict__ out_perp) {
  __shared__ float wsum[8];
  float h = 0.f;
  for (int m = threadIdx.x; m < MCODES; m += 256) {
    float avg = counts[m] * (1.0f / (float)NROWS);
    h += avg * logf(avg + EPSV);
  }
#pragma unroll
  for (int off = 16; off >= 1; off >>= 1) h += __shfl_xor(h, off, 32);
  const int lane = threadIdx.x & 31, wv = threadIdx.x >> 5;
  if (lane == 0) wsum[wv] = h;
  __syncthreads();
  if (threadIdx.x == 0) {
    float tot = 0.f;
#pragma unroll
    for (int i = 0; i < 8; ++i) tot += wsum[i];
    *out_perp = expf(-tot);
    *out_loss = (COMMIT + 1.0f) * (*sumsq) / (float)((size_t)NROWS * DDIM);
  }
}

extern "C" void kernel_launch(void* const* d_in, const int* in_sizes, int n_in,
                              void* d_out, int out_size, void* d_ws, size_t ws_size,
                              hipStream_t stream) {
  const float* x   = (const float*)d_in[0];   // [8,4096,256]
  const float* emb = (const float*)d_in[1];   // [1024,256]

  float* out      = (float*)d_out;
  float* q_out    = out;                            // N*D floats
  float* loss_out = out + (size_t)NROWS * DDIM;     // 1
  float* idxf_out = loss_out + 1;                   // N floats (indices as f32)
  float* perp_out = idxf_out + NROWS;               // 1

  char*  ws     = (char*)d_ws;
  int*   idx_i  = (int*)ws;                                  // N ints      (128 KB)
  float* enormv = (float*)(ws + (size_t)NROWS * sizeof(int));// M floats    (4 KB)
  float* counts = enormv + MCODES;                           // M floats    (4 KB)
  float* sumsq  = counts + MCODES;                           // 1 float
  _Float16* ehi = (_Float16*)(sumsq + 4);                    // M*D f16     (512 KB)
  _Float16* elo = ehi + (size_t)MCODES * DDIM;               // M*D f16     (512 KB)

  init_kernel<<<4, 256, 0, stream>>>(counts, sumsq);
  enorm_kernel<<<4, 256, 0, stream>>>(emb, enormv);
  emb_cvt_kernel<<<(MCODES * DDIM) / 256, 256, 0, stream>>>(emb, ehi, elo);
  vq_argmin_kernel<<<(NROWS / 16) / WAVES_PER_BLOCK, 32 * WAVES_PER_BLOCK, 0, stream>>>(
      x, ehi, elo, enormv, idx_i, idxf_out);
  vq_gather_kernel<<<NROWS / 4, 256, 0, stream>>>(x, emb, idx_i, q_out, counts, sumsq);
  vq_finalize_kernel<<<1, 256, 0, stream>>>(counts, sumsq, loss_out, perp_out);
}
